// MultiHeadAttention_49907519979821
// MI455X (gfx1250) — matrix operand order repaired and checked by reading, not compiled, not run on hardware
//
#include <hip/hip_runtime.h>
#include <hip/hip_bf16.h>

#define S_LEN   4096
#define DMODEL  1024
#define NHEAD   16
#define HDIM    64

typedef __bf16 bf16;
typedef __attribute__((ext_vector_type(16))) __bf16 v16bf;
typedef __attribute__((ext_vector_type(8)))  __bf16 v8bf;
typedef __attribute__((ext_vector_type(8)))  float  v8f;
typedef __attribute__((ext_vector_type(4)))  float  v4f;
typedef int v4i __attribute__((vector_size(16)));   // matches builtin's b128 unit

#if defined(__gfx1250__) && __has_builtin(__builtin_amdgcn_global_load_async_to_lds_b128)
#define HAVE_ASYNC_LDS 1
#else
#define HAVE_ASYNC_LDS 0
#endif

static __device__ inline v8f wmma_bf16(v16bf a, v16bf b, v8f c) {
  // (neg_a, A, neg_b, B, c_mod, C, reuse_a, reuse_b)
  return __builtin_amdgcn_wmma_f32_16x16x32_bf16(false, a, false, b, (short)0, c, false, false);
}

static __device__ inline v16bf cat8(v8bf lo, v8bf hi) {
  return __builtin_shufflevector(lo, hi, 0,1,2,3,4,5,6,7,8,9,10,11,12,13,14,15);
}

static __device__ inline v8bf cvt8(const float* __restrict__ s) {
  v4f a = *(const v4f*)s;
  v4f b = *(const v4f*)(s + 4);
  v8bf o;
  o[0]=(bf16)a[0]; o[1]=(bf16)a[1]; o[2]=(bf16)a[2]; o[3]=(bf16)a[3];
  o[4]=(bf16)b[0]; o[5]=(bf16)b[1]; o[6]=(bf16)b[2]; o[7]=(bf16)b[3];
  return o;
}

// Row reductions across one 16-lane half (xor masks <16 never cross halves).
static __device__ inline float hmax16(float x) {
  #pragma unroll
  for (int m = 8; m >= 1; m >>= 1) x = fmaxf(x, __shfl_xor(x, m, 32));
  return x;
}
static __device__ inline float hsum16(float x) {
  #pragma unroll
  for (int m = 8; m >= 1; m >>= 1) x += __shfl_xor(x, m, 32);
  return x;
}

// Wait until at most N async (ASYNCcnt) ops remain outstanding.
template<int N>
static __device__ inline void wait_async() {
#if HAVE_ASYNC_LDS
#if __has_builtin(__builtin_amdgcn_s_wait_asynccnt)
  __builtin_amdgcn_s_wait_asynccnt(N);
#else
  asm volatile("s_wait_asynccnt %0" :: "i"(N) : "memory");
#endif
#endif
}

// ---------------------------------------------------------------------------
// Tiled GEMM:  Y = X @ W^T + bias       X:[S,DMODEL]  W:[DMODEL,DMODEL] (fp32)
// BM=128 BN=128 BK=32, 256 threads = 8 waves in a 4(M) x 2(N) grid,
// each wave computes a 32x64 sub-tile as 2x4 WMMA fragments
// (8 WMMA on 2 A + 4 B LDS fragments per K-step).
// Sync staging is intentional here: fp32 -> bf16 conversion must pass
// through VGPRs, so async-to-LDS cannot be used for these operands.
//   XBF16:    X is bf16 (else fp32, converted on load)
//   TRANSOUT: store Y transposed as Y[n*S + m] (bf16) -- used for V
//   OUTF32:   store fp32 (final output projection), else bf16
// ---------------------------------------------------------------------------
template<bool XBF16, bool TRANSOUT, bool OUTF32>
__global__ __launch_bounds__(256)
void gemm_bias_kernel(const void* __restrict__ Xv, const float* __restrict__ W,
                      const float* __restrict__ bias, void* __restrict__ Yv) {
  constexpr int BM = 128, BN = 128, BK = 32;
  constexpr int AK = BK + 8;                 // pad: rows stay 16B aligned, shift banks
  __shared__ __align__(32) bf16 As[BM * AK];
  __shared__ __align__(32) bf16 Bs[BN * AK];

  const int tid  = threadIdx.x;
  const int lane = tid & 31, w = tid >> 5;
  const int half = lane >> 4, l16 = lane & 15;
  const int wm = w & 3, wn = w >> 2;
  const int m0 = blockIdx.y * BM;
  const int n0 = blockIdx.x * BN;

  const float* Xf = (const float*)Xv;
  const bf16*  Xb = (const bf16*)Xv;

  v8f c[2][4] = {};

  for (int k0 = 0; k0 < DMODEL; k0 += BK) {
    // ---- stage A tile: 128 rows x 32 k (each thread: 1 row-half = 16 elems)
    {
      const int row = tid >> 1, cg = (tid & 1) * 16;
      bf16* dst = &As[row * AK + cg];
      if (XBF16) {
        const bf16* src = Xb + (size_t)(m0 + row) * DMODEL + k0 + cg;
        *(v8bf*)dst       = *(const v8bf*)src;
        *(v8bf*)(dst + 8) = *(const v8bf*)(src + 8);
        if (k0 + BK < DMODEL) __builtin_prefetch(src + BK, 0, 0);
      } else {
        const float* src = Xf + (size_t)(m0 + row) * DMODEL + k0 + cg;
        *(v8bf*)dst       = cvt8(src);
        *(v8bf*)(dst + 8) = cvt8(src + 8);
        if (k0 + BK < DMODEL) __builtin_prefetch(src + BK, 0, 0);
      }
    }
    // ---- stage B tile: 128 N-rows x 32 k (each thread: 16 elems)
    {
      const int n = tid >> 1, cg = (tid & 1) * 16;
      const float* src = W + (size_t)(n0 + n) * DMODEL + k0 + cg;
      bf16* dst = &Bs[n * AK + cg];
      *(v8bf*)dst       = cvt8(src);
      *(v8bf*)(dst + 8) = cvt8(src + 8);
      if (k0 + BK < DMODEL) __builtin_prefetch(src + BK, 0, 0);
    }
    __syncthreads();

    #pragma unroll
    for (int fm = 0; fm < 2; ++fm) {
      // A frag: lane row = base + l16; chunks K = 8*half and 16 + 8*half
      const bf16* ap = &As[(wm * 32 + fm * 16 + l16) * AK + 8 * half];
      v16bf a = cat8(*(const v8bf*)ap, *(const v8bf*)(ap + 16));
      #pragma unroll
      for (int fn = 0; fn < 4; ++fn) {
        // B frag: lane col = base + l16; chunks K = 8*half and 16 + 8*half
        const bf16* bp = &Bs[(wn * 64 + fn * 16 + l16) * AK + 8 * half];
        v16bf b = cat8(*(const v8bf*)bp, *(const v8bf*)(bp + 16));
        c[fm][fn] = wmma_bf16(a, b, c[fm][fn]);
      }
    }
    __syncthreads();
  }

  // ---- epilogue: C-frag element r -> row r + 8*half, col l16
  #pragma unroll
  for (int fm = 0; fm < 2; ++fm) {
    #pragma unroll
    for (int fn = 0; fn < 4; ++fn) {
      const int col = n0 + wn * 64 + fn * 16 + l16;
      const float bv = bias[col];
      #pragma unroll
      for (int r = 0; r < 8; ++r) {
        const int row = m0 + wm * 32 + fm * 16 + r + 8 * half;
        const float val = c[fm][fn][r] + bv;
        if (OUTF32)         ((float*)Yv)[(size_t)row * DMODEL + col] = val;
        else if (TRANSOUT)  ((bf16*)Yv)[(size_t)col * S_LEN + row]   = (bf16)val;
        else                ((bf16*)Yv)[(size_t)row * DMODEL + col]  = (bf16)val;
      }
    }
  }
}

// ---------------------------------------------------------------------------
// Flash attention: one block = (head h, 128 query rows); 8 waves x 16 q-rows.
// 64-key tiles, double-buffered in LDS. Tiles are DMA'd global->LDS with
// GLOBAL_LOAD_ASYNC_TO_LDS_B128 (ASYNCcnt) with the next tile in flight while
// the current one feeds WMMA; sync staging fallback keeps the same pipeline.
// Qb,Kb: [S, DMODEL] bf16.  Vt: [DMODEL, S] bf16.  Ctx: [S, DMODEL] bf16.
// ---------------------------------------------------------------------------
static constexpr int KT  = 64;
static constexpr int KTP = KT + 8;

static __device__ inline void stage_tiles(const bf16* __restrict__ Kb,
                                          const bf16* __restrict__ Vt,
                                          int h, int kt,
                                          bf16* KtBuf, bf16* VtsBuf, int tid) {
  const int row = tid >> 2, cg = (tid & 3) * 16;
  const bf16* ks = Kb + (size_t)(kt + row) * DMODEL + h * HDIM + cg;
  const bf16* vs = Vt + (size_t)(h * HDIM + row) * S_LEN + kt + cg;
  bf16* kd = &KtBuf[row * KTP + cg];
  bf16* vd = &VtsBuf[row * KTP + cg];
#if HAVE_ASYNC_LDS
  typedef __attribute__((address_space(1))) v4i* gp128;   // global int4*
  typedef __attribute__((address_space(3))) v4i* lp128;   // LDS int4*
  __builtin_amdgcn_global_load_async_to_lds_b128((gp128)ks,       (lp128)kd,       0, 0);
  __builtin_amdgcn_global_load_async_to_lds_b128((gp128)(ks + 8), (lp128)(kd + 8), 0, 0);
  __builtin_amdgcn_global_load_async_to_lds_b128((gp128)vs,       (lp128)vd,       0, 0);
  __builtin_amdgcn_global_load_async_to_lds_b128((gp128)(vs + 8), (lp128)(vd + 8), 0, 0);
#else
  *(v8bf*)kd       = *(const v8bf*)ks;
  *(v8bf*)(kd + 8) = *(const v8bf*)(ks + 8);
  *(v8bf*)vd       = *(const v8bf*)vs;
  *(v8bf*)(vd + 8) = *(const v8bf*)(vs + 8);
#endif
}

__global__ __launch_bounds__(256)
void attn_kernel(const bf16* __restrict__ Qb, const bf16* __restrict__ Kb,
                 const bf16* __restrict__ Vt, bf16* __restrict__ Ctx) {
  __shared__ __align__(32) bf16 Kt [2 * KT   * KTP];   // [buf][key][d]
  __shared__ __align__(32) bf16 Vts[2 * HDIM * KTP];   // [buf][d][key]
  __shared__ __align__(32) bf16 Plds[8 * 16 * KTP];    // per-wave 16 x KT staging

  const int tid  = threadIdx.x;
  const int lane = tid & 31, w = tid >> 5;
  const int half = lane >> 4, l16 = lane & 15;
  const int h  = blockIdx.y;
  const int q0 = blockIdx.x * 128 + w * 16;

  // Q fragments (K-dim = head dim 64 -> 2 frags), loaded once from global.
  v16bf qf[2];
  #pragma unroll
  for (int f = 0; f < 2; ++f) {
    const bf16* qp = Qb + (size_t)(q0 + l16) * DMODEL + h * HDIM + f * 32 + 8 * half;
    qf[f] = cat8(*(const v8bf*)qp, *(const v8bf*)(qp + 16));
  }

  v8f acc[4] = {};
  float mrow[8], lrow[8];
  #pragma unroll
  for (int r = 0; r < 8; ++r) { mrow[r] = -__builtin_inff(); lrow[r] = 0.0f; }

  bf16* Pw = &Plds[w * 16 * KTP];

  // Prologue: start tile 0 into buffer 0.
  stage_tiles(Kb, Vt, h, 0, &Kt[0], &Vts[0], tid);

  for (int kt = 0; kt < S_LEN; kt += KT) {
    const int cur = (kt >> 6) & 1;
    bf16* KtC  = &Kt [cur        * KT   * KTP];
    bf16* VtsC = &Vts[cur        * HDIM * KTP];
    bf16* KtN  = &Kt [(cur ^ 1)  * KT   * KTP];
    bf16* VtsN = &Vts[(cur ^ 1)  * HDIM * KTP];

    if (kt + KT < S_LEN) {
      stage_tiles(Kb, Vt, h, kt + KT, KtN, VtsN, tid);   // next tile in flight
      wait_async<4>();   // current tile's 4 per-thread loads retired (in-order)
    } else {
      wait_async<0>();
    }
    __syncthreads();

    // ---- S = Q K^T : 16 x 64 as 4 column fragments
    v8f sf[4] = {};
    #pragma unroll
    for (int f = 0; f < 2; ++f) {
      #pragma unroll
      for (int j = 0; j < 4; ++j) {
        const bf16* bp = &KtC[(j * 16 + l16) * KTP + f * 32 + 8 * half];
        v16bf b = cat8(*(const v8bf*)bp, *(const v8bf*)(bp + 16));
        sf[j] = wmma_bf16(qf[f], b, sf[j]);
      }
    }
    #pragma unroll
    for (int j = 0; j < 4; ++j) sf[j] *= 0.125f;   // 1/sqrt(64)

    // ---- online softmax
    float mnew[8], alpha[8];
    #pragma unroll
    for (int r = 0; r < 8; ++r) {
      float mx = fmaxf(fmaxf(sf[0][r], sf[1][r]), fmaxf(sf[2][r], sf[3][r]));
      mx = hmax16(mx);
      const float mn = fmaxf(mrow[r], mx);
      mnew[r]  = mn;
      alpha[r] = __expf(mrow[r] - mn);
    }
    #pragma unroll
    for (int j = 0; j < 4; ++j)
      #pragma unroll
      for (int r = 0; r < 8; ++r)
        sf[j][r] = __expf(sf[j][r] - mnew[r]);
    #pragma unroll
    for (int r = 0; r < 8; ++r) {
      float sm = sf[0][r] + sf[1][r] + sf[2][r] + sf[3][r];
      sm = hsum16(sm);
      lrow[r] = lrow[r] * alpha[r] + sm;
      mrow[r] = mnew[r];
    }
    #pragma unroll
    for (int j = 0; j < 4; ++j)
      #pragma unroll
      for (int r = 0; r < 8; ++r)
        acc[j][r] *= alpha[r];

    // ---- stage P (bf16, row-major 16 x KT) to this wave's LDS region
    #pragma unroll
    for (int j = 0; j < 4; ++j)
      #pragma unroll
      for (int r = 0; r < 8; ++r)
        Pw[(r + 8 * half) * KTP + j * 16 + l16] = (bf16)sf[j][r];

    // ---- ctx += P @ Vtile  (A = P rows, B = Vt columns, both contiguous)
    #pragma unroll
    for (int f = 0; f < 2; ++f) {
      const bf16* pp = &Pw[l16 * KTP + f * 32 + 8 * half];
      v16bf pa = cat8(*(const v8bf*)pp, *(const v8bf*)(pp + 16));
      #pragma unroll
      for (int j = 0; j < 4; ++j) {
        const bf16* vp = &VtsC[(j * 16 + l16) * KTP + f * 32 + 8 * half];
        v16bf b = cat8(*(const v8bf*)vp, *(const v8bf*)(vp + 16));
        acc[j] = wmma_bf16(pa, b, acc[j]);
      }
    }
    __syncthreads();   // all waves done with buffer `cur` before it is re-staged
  }

  // ---- normalize and write ctx (concat-head layout [S, DMODEL])
  #pragma unroll
  for (int r = 0; r < 8; ++r) {
    const float inv = 1.0f / lrow[r];
    const int row = q0 + r + 8 * half;
    #pragma unroll
    for (int j = 0; j < 4; ++j)
      Ctx[(size_t)row * DMODEL + h * HDIM + j * 16 + l16] = (bf16)(acc[j][r] * inv);
  }
}

// ---------------------------------------------------------------------------
extern "C" void kernel_launch(void* const* d_in, const int* in_sizes, int n_in,
                              void* d_out, int out_size, void* d_ws, size_t ws_size,
                              hipStream_t stream) {
  (void)in_sizes; (void)n_in; (void)out_size; (void)ws_size;
  const float* q   = (const float*)d_in[0];
  const float* k   = (const float*)d_in[1];
  const float* v   = (const float*)d_in[2];
  const float* w_q = (const float*)d_in[3];
  const float* b_q = (const float*)d_in[4];
  const float* w_k = (const float*)d_in[5];
  const float* b_k = (const float*)d_in[6];
  const float* w_v = (const float*)d_in[7];
  const float* b_v = (const float*)d_in[8];
  const float* w_o = (const float*)d_in[9];
  const float* b_o = (const float*)d_in[10];

  const size_t SD = (size_t)S_LEN * DMODEL;
  bf16* Qb  = (bf16*)d_ws;       // [S, D]
  bf16* Kb  = Qb + SD;           // [S, D]
  bf16* Vtr = Kb + SD;           // [D, S]  (V transposed)
  bf16* Ctx = Vtr + SD;          // [S, D]

  dim3 gg(DMODEL / 128, S_LEN / 128);
  gemm_bias_kernel<false, false, false><<<gg, 256, 0, stream>>>(q, w_q, b_q, Qb);
  gemm_bias_kernel<false, false, false><<<gg, 256, 0, stream>>>(k, w_k, b_k, Kb);
  gemm_bias_kernel<false, true,  false><<<gg, 256, 0, stream>>>(v, w_v, b_v, Vtr);

  attn_kernel<<<dim3(S_LEN / 128, NHEAD), 256, 0, stream>>>(Qb, Kb, Vtr, Ctx);

  gemm_bias_kernel<true, false, true><<<gg, 256, 0, stream>>>(Ctx, w_o, b_o, d_out);
}
